// CausalSelfAttention_18726057410992
// MI455X (gfx1250) — matrix operand order repaired and checked by reading, not compiled, not run on hardware
//
#include <hip/hip_runtime.h>
#include <hip/hip_bf16.h>
#include <math.h>

typedef __attribute__((ext_vector_type(16))) __bf16 v16bf;
typedef __attribute__((ext_vector_type(8)))  __bf16 v8bf;
typedef __attribute__((ext_vector_type(8)))  float  v8f;

#define B_    4
#define T_    2048
#define C_    1024
#define H_    16
#define D_    64
#define M_TOT (B_ * T_)     // 8192
#define N_QKV (3 * C_)      // 3072

// Fragment-native B-tile storage geometry (uints):
//   entry [tile][lane][j] = pair {B[kh+2j][n], B[kh+2j+1][n]},
//   lane = (n&15) + 16*(k>=16), j = 0..7.
// inner lane stride 12 uints (48B: 16B-aligned b128 reads),
// tile stride 388 uints (== 4 mod 64 banks: conflict-free staging stores).
#define BF_LANE_STRIDE 12
#define BF_TILE_STRIDE 388

// ---------------------------------------------------------------------------
// Helpers
// ---------------------------------------------------------------------------
__device__ __forceinline__ unsigned pack2bf(float a, float b) {
  union { __bf16 h[2]; unsigned u; } x;
  x.h[0] = (__bf16)a;
  x.h[1] = (__bf16)b;
  return x.u;
}

// A-fragment (16x32 bf16): per-lane K-contiguous halves -> two ds_load_b128.
__device__ __forceinline__ v16bf load_a_frag(const __bf16* base, int stride,
                                             int mbase, int kbase, int lane) {
  const int m  = mbase + (lane & 15);
  const int kh = (lane >> 4) << 4;  // 0 or 16
  const __bf16* p = base + m * stride + kbase + kh;
  v8bf lo = *(const v8bf*)p;
  v8bf hi = *(const v8bf*)(p + 8);
  return __builtin_shufflevector(lo, hi, 0, 1, 2, 3, 4, 5, 6, 7,
                                 8, 9, 10, 11, 12, 13, 14, 15);
}

// B-fragment from fragment-native tile: per-lane 16 bf16 contiguous ->
// two ds_load_b128, zero packing VALU.
__device__ __forceinline__ v16bf load_b_frag_fn(const unsigned* base, int tile,
                                                int lane) {
  const __bf16* p =
      (const __bf16*)(base + tile * BF_TILE_STRIDE + lane * BF_LANE_STRIDE);
  v8bf lo = *(const v8bf*)p;
  v8bf hi = *(const v8bf*)(p + 8);
  return __builtin_shufflevector(lo, hi, 0, 1, 2, 3, 4, 5, 6, 7,
                                 8, 9, 10, 11, 12, 13, 14, 15);
}

#define WMMA_BF16(A, B, C) \
  __builtin_amdgcn_wmma_f32_16x16x32_bf16(false, (A), false, (B), (short)0, (C), false, false)

// ---------------------------------------------------------------------------
// Kernel 1: qkv = x @ w_qkv + b_qkv, scattered to Q/K/V [B,H,T,D] in bf16.
// Q is pre-scaled by 1/sqrt(D) so the attention kernel skips per-score scaling.
// Block tile 128x128, K-chunk 64, 8 waves as 4(M)x2(N), wave tile 32x64.
// ---------------------------------------------------------------------------
__global__ __launch_bounds__(256)
void qkv_gemm_kernel(const float* __restrict__ x, const float* __restrict__ w,
                     const float* __restrict__ bias,
                     __bf16* __restrict__ qb, __bf16* __restrict__ kb,
                     __bf16* __restrict__ vb) {
  __shared__ __bf16   As[128][72];
  __shared__ unsigned Bf[16 * BF_TILE_STRIDE];
  const int tid  = threadIdx.x;
  const int lane = tid & 31;
  const int wave = tid >> 5;
  const int wm   = wave >> 1;
  const int wn   = wave & 1;
  const int m0   = blockIdx.y * 128;
  const int n0   = blockIdx.x * 128;

  v8f zero = {};
  v8f acc[2][4];
#pragma unroll
  for (int i = 0; i < 2; ++i)
#pragma unroll
    for (int j = 0; j < 4; ++j) acc[i][j] = zero;

  const int k2    = tid >> 3;         // pair-row (k = 2*k2, 2*k2+1)
  const int cb8   = (tid & 7) << 4;   // 16 columns
  const int bs    = k2 >> 4;          // K-subtile
  const int khalf = (k2 >> 3) & 1;
  const int bj    = k2 & 7;           // pair index inside tile

  for (int kk = 0; kk < C_; kk += 64) {
    {  // stage A: 128x64 fp32 -> bf16
      const int r  = tid >> 1;
      const int cc = (tid & 1) << 5;
      const float4* src = (const float4*)(x + (size_t)(m0 + r) * C_ + kk + cc);
      unsigned* dst = (unsigned*)&As[r][cc];
#pragma unroll
      for (int i = 0; i < 8; ++i) {
        float4 f = src[i];
        dst[2 * i]     = pack2bf(f.x, f.y);
        dst[2 * i + 1] = pack2bf(f.z, f.w);
      }
    }
    {  // stage B: 64x128 fp32 -> fragment-native pairs
      const float4* s0 = (const float4*)(w + (size_t)(kk + 2 * k2) * N_QKV + n0 + cb8);
      const float4* s1 = (const float4*)(w + (size_t)(kk + 2 * k2 + 1) * N_QKV + n0 + cb8);
      unsigned* dst = Bf + (bs * 8 + (cb8 >> 4)) * BF_TILE_STRIDE + bj;
#pragma unroll
      for (int i = 0; i < 4; ++i) {
        float4 f0 = s0[i], f1 = s1[i];
        dst[(4 * i + 0 + 16 * khalf) * BF_LANE_STRIDE] = pack2bf(f0.x, f1.x);
        dst[(4 * i + 1 + 16 * khalf) * BF_LANE_STRIDE] = pack2bf(f0.y, f1.y);
        dst[(4 * i + 2 + 16 * khalf) * BF_LANE_STRIDE] = pack2bf(f0.z, f1.z);
        dst[(4 * i + 3 + 16 * khalf) * BF_LANE_STRIDE] = pack2bf(f0.w, f1.w);
      }
    }
    __syncthreads();
#pragma unroll
    for (int ks = 0; ks < 64; ks += 32) {
      v16bf a0 = load_a_frag(&As[0][0], 72, wm * 32, ks, lane);
      v16bf a1 = load_a_frag(&As[0][0], 72, wm * 32 + 16, ks, lane);
#pragma unroll
      for (int j = 0; j < 4; ++j) {
        v16bf b = load_b_frag_fn(Bf, (ks >> 5) * 8 + wn * 4 + j, lane);
        acc[0][j] = WMMA_BF16(a0, b, acc[0][j]);
        acc[1][j] = WMMA_BF16(a1, b, acc[1][j]);
      }
    }
    __syncthreads();
  }

  // Epilogue: bias add (+1/sqrt(D) prescale for Q) + scatter to [B,H,T,D] bf16
  const int hi = lane >> 4, nlo = lane & 15;
#pragma unroll
  for (int j = 0; j < 4; ++j) {
    const int n  = n0 + wn * 64 + j * 16 + nlo;
    const float bv = bias[n];
    const int which = n >> 10;  // 0=q 1=k 2=v
    const float scl = (which == 0) ? 0.125f : 1.0f;
    const int cc = n & (C_ - 1);
    const int h  = cc >> 6, d = cc & 63;
    __bf16* dst = (which == 0) ? qb : (which == 1) ? kb : vb;
#pragma unroll
    for (int i = 0; i < 2; ++i) {
#pragma unroll
      for (int r = 0; r < 8; ++r) {
        const int m  = m0 + wm * 32 + i * 16 + r + 8 * hi;
        const int bb = m >> 11, t = m & (T_ - 1);
        dst[(((size_t)(bb * H_ + h)) * T_ + t) * D_ + d] =
            (__bf16)((acc[i][j][r] + bv) * scl);
      }
    }
  }
}

// ---------------------------------------------------------------------------
// Kernel 2: flash-attention. Block = (b,h) x 128 q-rows; 8 waves x 16 rows.
// q-tiles mapped in reverse so the longest (most k-tiles) blocks launch first.
// ---------------------------------------------------------------------------
__global__ __launch_bounds__(256)
void attn_kernel(const __bf16* __restrict__ q, const __bf16* __restrict__ k,
                 const __bf16* __restrict__ v, __bf16* __restrict__ y) {
  __shared__ __bf16   Qs[128][72];
  __shared__ unsigned KBf[8 * BF_TILE_STRIDE];
  __shared__ unsigned VBf[8 * BF_TILE_STRIDE];
  __shared__ __bf16   Ps[8][16][72];
  const int tid  = threadIdx.x;
  const int lane = tid & 31;
  const int wave = tid >> 5;
  const int bh   = blockIdx.y;
  const int q0   = (gridDim.x - 1 - blockIdx.x) * 128;  // heavy tiles first
  const __bf16* Qp = q + (size_t)bh * T_ * D_;
  const __bf16* Kp = k + (size_t)bh * T_ * D_;
  const __bf16* Vp = v + (size_t)bh * T_ * D_;

  {  // stage Q tile 128x64 (bf16 copy)
    const int r  = tid >> 1;
    const int cc = (tid & 1) << 5;
    const unsigned* src = (const unsigned*)(Qp + (size_t)(q0 + r) * D_ + cc);
    unsigned* dst = (unsigned*)&Qs[r][cc];
#pragma unroll
    for (int i = 0; i < 16; ++i) dst[i] = src[i];
  }
  __syncthreads();

  const int m0 = wave * 16;
  v16bf aq0 = load_a_frag(&Qs[0][0], 72, m0, 0, lane);
  v16bf aq1 = load_a_frag(&Qs[0][0], 72, m0, 32, lane);

  v8f zero = {};
  v8f o[4];
  float mrow[8], lrow[8];
#pragma unroll
  for (int n = 0; n < 4; ++n) o[n] = zero;
#pragma unroll
  for (int r = 0; r < 8; ++r) { mrow[r] = -1e30f; lrow[r] = 0.0f; }

  const int hi = lane >> 4, nlo = lane & 15;
  const int row_base = q0 + m0 + 8 * hi;

  // K staging decomposition: thread reads kcol c, 8 d-pairs starting at ab
  const int kc    = tid >> 2;
  const int kab   = (tid & 3) << 3;
  const int kS    = kab >> 4;
  const int kHalf = (kab >> 3) & 1;
  const int kLane = (kc & 15) + 16 * kHalf;
  const int kNt   = kc >> 4;
  // V staging decomposition: kcol pair c2, 8 d's starting at db
  const int vc2   = tid >> 3;
  const int vdb   = (tid & 7) << 3;
  const int vS    = vc2 >> 4;
  const int vHalf = (vc2 >> 3) & 1;
  const int vJ    = vc2 & 7;
  const int vNt   = vdb >> 4;
  const int vDlo  = vdb & 15;

  const int jmax = (q0 >> 6) + 2;  // causal tile bound
  for (int j = 0; j < jmax; ++j) {
    const int kb0 = j * 64;
    {  // stage K^T fragment-native: contiguous 8-uint (32B) write per thread
      const unsigned* ks = (const unsigned*)(Kp + (size_t)(kb0 + kc) * D_ + 2 * kab);
      unsigned* dst = KBf + (kS * 4 + kNt) * BF_TILE_STRIDE + kLane * BF_LANE_STRIDE;
#pragma unroll
      for (int i = 0; i < 8; ++i) dst[i] = ks[i];
    }
    {  // stage V fragment-native: interleave rows via v_perm
      const unsigned* a0 = (const unsigned*)(Vp + (size_t)(kb0 + 2 * vc2) * D_ + vdb);
      const unsigned* a1 = (const unsigned*)(Vp + (size_t)(kb0 + 2 * vc2 + 1) * D_ + vdb);
      unsigned* dst = VBf + (vS * 4 + vNt) * BF_TILE_STRIDE + vJ;
#pragma unroll
      for (int i = 0; i < 4; ++i) {
        unsigned x0 = a0[i], x1 = a1[i];
        dst[(vDlo + 2 * i + 0 + 16 * vHalf) * BF_LANE_STRIDE] =
            __builtin_amdgcn_perm(x1, x0, 0x05040100u);
        dst[(vDlo + 2 * i + 1 + 16 * vHalf) * BF_LANE_STRIDE] =
            __builtin_amdgcn_perm(x1, x0, 0x07060302u);
      }
    }
    __syncthreads();

    // S = Q @ K^T (16x64 per wave); Q is pre-scaled by 1/sqrt(D)
    v8f s[4];
#pragma unroll
    for (int t = 0; t < 4; ++t) {
      v16bf bk0 = load_b_frag_fn(KBf, 0 * 4 + t, lane);
      v16bf bk1 = load_b_frag_fn(KBf, 1 * 4 + t, lane);
      s[t] = zero;
      s[t] = WMMA_BF16(aq0, bk0, s[t]);
      s[t] = WMMA_BF16(aq1, bk1, s[t]);
    }

    // Causal mask only needed for tiles crossing this wave's diagonal band
    // (wave-uniform branch: EXEC stays all-1s).
    if (kb0 + 63 > q0 + m0) {
#pragma unroll
      for (int t = 0; t < 4; ++t) {
        const int col = kb0 + t * 16 + nlo;
#pragma unroll
        for (int r = 0; r < 8; ++r)
          s[t][r] = (col <= row_base + r) ? s[t][r] : -1e30f;
      }
    }

    // Online softmax over 64 columns
#pragma unroll
    for (int r = 0; r < 8; ++r) {
      float vmax = fmaxf(fmaxf(s[0][r], s[1][r]), fmaxf(s[2][r], s[3][r]));
#pragma unroll
      for (int off = 1; off < 16; off <<= 1)
        vmax = fmaxf(vmax, __shfl_xor(vmax, off, 32));
      const float mnew  = fmaxf(mrow[r], vmax);
      const float alpha = __expf(mrow[r] - mnew);
      float psum = 0.0f;
#pragma unroll
      for (int t = 0; t < 4; ++t) {
        const float p = __expf(s[t][r] - mnew);
        psum += p;
        Ps[wave][r + 8 * hi][t * 16 + nlo] = (__bf16)p;
      }
#pragma unroll
      for (int off = 1; off < 16; off <<= 1) psum += __shfl_xor(psum, off, 32);
      lrow[r] = lrow[r] * alpha + psum;
      mrow[r] = mnew;
#pragma unroll
      for (int n = 0; n < 4; ++n) o[n][r] *= alpha;
    }

    // O += P @ V (DS ops in-order within wave: Ps writes precede reads)
    v16bf ap0 = load_a_frag(&Ps[wave][0][0], 72, 0, 0, lane);
    v16bf ap1 = load_a_frag(&Ps[wave][0][0], 72, 0, 32, lane);
#pragma unroll
    for (int n = 0; n < 4; ++n) {
      v16bf bv0 = load_b_frag_fn(VBf, 0 * 4 + n, lane);
      v16bf bv1 = load_b_frag_fn(VBf, 1 * 4 + n, lane);
      o[n] = WMMA_BF16(ap0, bv0, o[n]);
      o[n] = WMMA_BF16(ap1, bv1, o[n]);
    }
    __syncthreads();
  }

  // Epilogue: y[B,T,C] bf16, un-permuting heads (8 reciprocals, then muls)
  const int bb = bh >> 4, h = bh & 15;
  float linv[8];
#pragma unroll
  for (int r = 0; r < 8; ++r) linv[r] = 1.0f / lrow[r];
#pragma unroll
  for (int n = 0; n < 4; ++n) {
#pragma unroll
    for (int r = 0; r < 8; ++r) {
      const int row = row_base + r;
      const int d   = n * 16 + nlo;
      y[((size_t)(bb * T_ + row)) * C_ + h * D_ + d] = (__bf16)(o[n][r] * linv[r]);
    }
  }
}

// ---------------------------------------------------------------------------
// Kernel 3: out = y @ w_proj + b_proj (fp32 output)
// ---------------------------------------------------------------------------
__global__ __launch_bounds__(256)
void proj_gemm_kernel(const __bf16* __restrict__ yb, const float* __restrict__ w,
                      const float* __restrict__ bias, float* __restrict__ out) {
  __shared__ __bf16   As[128][72];
  __shared__ unsigned Bf[16 * BF_TILE_STRIDE];
  const int tid  = threadIdx.x;
  const int lane = tid & 31;
  const int wave = tid >> 5;
  const int wm   = wave >> 1;
  const int wn   = wave & 1;
  const int m0   = blockIdx.y * 128;
  const int n0   = blockIdx.x * 128;

  v8f zero = {};
  v8f acc[2][4];
#pragma unroll
  for (int i = 0; i < 2; ++i)
#pragma unroll
    for (int j = 0; j < 4; ++j) acc[i][j] = zero;

  const int k2    = tid >> 3;
  const int cb8   = (tid & 7) << 4;
  const int bs    = k2 >> 4;
  const int khalf = (k2 >> 3) & 1;
  const int bj    = k2 & 7;

  for (int kk = 0; kk < C_; kk += 64) {
    {  // stage A: bf16 copy
      const int r  = tid >> 1;
      const int cc = (tid & 1) << 5;
      const unsigned* src = (const unsigned*)(yb + (size_t)(m0 + r) * C_ + kk + cc);
      unsigned* dst = (unsigned*)&As[r][cc];
#pragma unroll
      for (int i = 0; i < 16; ++i) dst[i] = src[i];
    }
    {  // stage B: fp32 -> fragment-native pairs
      const float4* s0 = (const float4*)(w + (size_t)(kk + 2 * k2) * C_ + n0 + cb8);
      const float4* s1 = (const float4*)(w + (size_t)(kk + 2 * k2 + 1) * C_ + n0 + cb8);
      unsigned* dst = Bf + (bs * 8 + (cb8 >> 4)) * BF_TILE_STRIDE + bj;
#pragma unroll
      for (int i = 0; i < 4; ++i) {
        float4 f0 = s0[i], f1 = s1[i];
        dst[(4 * i + 0 + 16 * khalf) * BF_LANE_STRIDE] = pack2bf(f0.x, f1.x);
        dst[(4 * i + 1 + 16 * khalf) * BF_LANE_STRIDE] = pack2bf(f0.y, f1.y);
        dst[(4 * i + 2 + 16 * khalf) * BF_LANE_STRIDE] = pack2bf(f0.z, f1.z);
        dst[(4 * i + 3 + 16 * khalf) * BF_LANE_STRIDE] = pack2bf(f0.w, f1.w);
      }
    }
    __syncthreads();
#pragma unroll
    for (int ks = 0; ks < 64; ks += 32) {
      v16bf a0 = load_a_frag(&As[0][0], 72, wm * 32, ks, lane);
      v16bf a1 = load_a_frag(&As[0][0], 72, wm * 32 + 16, ks, lane);
#pragma unroll
      for (int j = 0; j < 4; ++j) {
        v16bf b = load_b_frag_fn(Bf, (ks >> 5) * 8 + wn * 4 + j, lane);
        acc[0][j] = WMMA_BF16(a0, b, acc[0][j]);
        acc[1][j] = WMMA_BF16(a1, b, acc[1][j]);
      }
    }
    __syncthreads();
  }

  const int hi = lane >> 4, nlo = lane & 15;
#pragma unroll
  for (int j = 0; j < 4; ++j) {
    const int n  = n0 + wn * 64 + j * 16 + nlo;
    const float bv = bias[n];
#pragma unroll
    for (int i = 0; i < 2; ++i) {
#pragma unroll
      for (int r = 0; r < 8; ++r) {
        const int m = m0 + wm * 32 + i * 16 + r + 8 * hi;
        out[(size_t)m * C_ + n] = acc[i][j][r] + bv;
      }
    }
  }
}

// ---------------------------------------------------------------------------
extern "C" void kernel_launch(void* const* d_in, const int* in_sizes, int n_in,
                              void* d_out, int out_size, void* d_ws, size_t ws_size,
                              hipStream_t stream) {
  (void)in_sizes; (void)n_in; (void)out_size; (void)ws_size;
  const float* x      = (const float*)d_in[0];
  const float* w_qkv  = (const float*)d_in[1];
  const float* b_qkv  = (const float*)d_in[2];
  const float* w_proj = (const float*)d_in[3];
  const float* b_proj = (const float*)d_in[4];
  float* out = (float*)d_out;

  char* ws = (char*)d_ws;
  size_t off = 0;
  auto take = [&](size_t bytes) -> char* {
    char* p = ws + off;
    off += (bytes + 255) & ~(size_t)255;
    return p;
  };
  const size_t qkv_elems = (size_t)B_ * H_ * T_ * D_;
  __bf16* qb = (__bf16*)take(qkv_elems * 2);
  __bf16* kb = (__bf16*)take(qkv_elems * 2);
  __bf16* vb = (__bf16*)take(qkv_elems * 2);
  __bf16* yb = (__bf16*)take((size_t)B_ * T_ * C_ * 2);

  qkv_gemm_kernel<<<dim3(N_QKV / 128, M_TOT / 128), 256, 0, stream>>>(
      x, w_qkv, b_qkv, qb, kb, vb);
  attn_kernel<<<dim3(T_ / 128, B_ * H_), 256, 0, stream>>>(qb, kb, vb, yb);
  proj_gemm_kernel<<<dim3(C_ / 128, M_TOT / 128), 256, 0, stream>>>(
      yb, w_proj, b_proj, out);
}
